// ResidualBlock_12352325943971
// MI455X (gfx1250) — compile-verified, matrix operand order checked by reading, not run on hardware
//
#include <hip/hip_runtime.h>
#include <hip/hip_fp16.h>

typedef __attribute__((ext_vector_type(16))) _Float16 v16h;
typedef __attribute__((ext_vector_type(8)))  float    v8f;
typedef __attribute__((ext_vector_type(4)))  float    f32x4;

#define EPSV  1e-5f
#define QLEVF 255.0f
#define WMAXF 127.0f

#define NB   32
#define CH   128
#define HH   56
#define WW2  56
#define HWSZ (HH*WW2)        // 3136
#define MTOT (NB*HWSZ)       // 100352 (== 784 * 128)
#define KSZ  9
#define NCHUNK (KSZ*4)       // 36 K-chunks of 32 input channels
#define LDSPITCH 40          // 32 f16 + 8 pad (80B rows, 16B aligned)

// ---------------- CDNA5 async global->LDS (16B per lane, ASYNCcnt-tracked) ----------------
__device__ __forceinline__ void async_load_b128_to_lds(unsigned lds_off, const void* gptr) {
  asm volatile("global_load_async_to_lds_b128 %0, %1, off"
               :: "v"(lds_off), "v"((unsigned long long)(uintptr_t)gptr)
               : "memory");
}
__device__ __forceinline__ void wait_asynccnt0() {
#if __has_builtin(__builtin_amdgcn_s_wait_asynccnt)
  __builtin_amdgcn_s_wait_asynccnt(0);
#else
  asm volatile("s_wait_asynccnt 0" ::: "memory");
#endif
}

// ---------------- weight prep: per-out-channel int8 fake-quant + OIHW -> [tap][o][c] f16
__global__ __launch_bounds__(128) void wprep_kernel(const float* __restrict__ w1,
                                                    const float* __restrict__ w2,
                                                    _Float16* __restrict__ w1t,
                                                    _Float16* __restrict__ w2t) {
  const int b = blockIdx.x;
  const int tensor = b >> 7;
  const int o = b & 127;
  const float* wsrc = (tensor ? w2 : w1) + (size_t)o * (CH * KSZ);
  _Float16* wdst = tensor ? w2t : w1t;
  const int t = threadIdx.x;

  __shared__ float red[128];
  float mx = 0.f;
  for (int i = t; i < CH * KSZ; i += 128) mx = fmaxf(mx, fabsf(wsrc[i]));
  red[t] = mx;
  __syncthreads();
  for (int s = 64; s > 0; s >>= 1) {
    if (t < s) red[t] = fmaxf(red[t], red[t + s]);
    __syncthreads();
  }
  const float sc = fmaxf(red[0] / WMAXF, 1e-8f);

  const int c = t;  // one input channel per thread
  #pragma unroll
  for (int kk = 0; kk < KSZ; ++kk) {
    float wv = wsrc[c * KSZ + kk];
    float q = rintf(wv / sc);
    q = fminf(fmaxf(q, -WMAXF), WMAXF) * sc;
    wdst[((size_t)kk * CH + o) * CH + c] = (_Float16)q;
  }
}

// ---------------- fused conv3x3 + BN + PACT (+ residual requant-add for stage 2)
union FragH { v16h v; f32x4 q[2]; };
union PackH { _Float16 h[16]; f32x4 q[2]; };

template <bool SECOND>
__device__ __forceinline__ void load_act_chunk(int i, int n0, int h0, int w0, int cgrp,
                                               const float* __restrict__ in_f32,
                                               const _Float16* __restrict__ in_f16,
                                               float out[16]) {
  const int kk = i >> 2, cc = i & 3;
  const int kr = kk / 3;
  const int dh = kr - 1, dw = (kk - kr * 3) - 1;
  const int h2 = h0 + dh, w2 = w0 + dw;
  const bool okb = (h2 >= 0) & (h2 < HH) & (w2 >= 0) & (w2 < WW2);
  const int sp = h2 * WW2 + w2;
  #pragma unroll
  for (int j = 0; j < 16; ++j) {
    const int cglob = cc * 32 + cgrp * 16 + j;   // 16 contiguous channels per thread
    float v = 0.f;
    if (okb) {  // uniform across all 16 loads of this thread -> single exec toggle
      const size_t idx = ((size_t)(n0 * CH + cglob)) * HWSZ + sp;
      v = SECOND ? (float)in_f16[idx] : in_f32[idx];
    }
    out[j] = v;
  }
}

template <bool SECOND>
__global__ __launch_bounds__(256) void conv_kernel(
    const float* __restrict__ in_f32,    // stage1 input x (f32)
    const _Float16* __restrict__ in_f16, // stage2 input (mid, f16)
    const _Float16* __restrict__ wt,     // [9][o=128][c=128] f16
    const float* __restrict__ gp, const float* __restrict__ bp,
    const float* __restrict__ mp, const float* __restrict__ vp,
    const float* __restrict__ ap,
    const float* __restrict__ a3p,       // SECOND only
    const float* __restrict__ xres,      // SECOND only
    _Float16* __restrict__ out_f16,      // stage1 output (mid)
    float* __restrict__ out_f32)         // stage2 output
{
  __shared__ __align__(16) _Float16 lds_wt [2][128 * LDSPITCH];
  __shared__ __align__(16) _Float16 lds_act[2][128 * LDSPITCH];
  __shared__ float p_scale[128], p_bias[128], p_alpha[128], p_qs[128], p_qinv[128];
  __shared__ float p3_alpha[128], p3_qs[128], p3_qinv[128];

  const int t = threadIdx.x;
  if (t < 128) {
    const float scv = gp[t] * rsqrtf(vp[t] + EPSV);
    p_scale[t] = scv;
    p_bias[t]  = bp[t] - mp[t] * scv;
    const float al = ap[t];
    p_alpha[t] = al; p_qs[t] = al / QLEVF; p_qinv[t] = QLEVF / al;
    if (SECOND) {
      const float a3 = a3p[t];
      p3_alpha[t] = a3; p3_qs[t] = a3 / QLEVF; p3_qinv[t] = QLEVF / a3;
    }
  }

  const int tile_base = blockIdx.x * 128;
  // per-thread activation-load coordinates (lanes sweep spatial -> coalesced)
  const int pos  = t & 127;
  const int cgrp = t >> 7;  // 0/1 : which 16-channel half this thread stages
  const int m0 = tile_base + pos;
  const int n0 = m0 / HWSZ;
  const int hw0 = m0 - n0 * HWSZ;
  const int h0 = hw0 / WW2;
  const int w0 = hw0 - h0 * WW2;

  const int lane  = t & 31;
  const int wv    = t >> 5;
  const int wm    = wv >> 1;             // 0..3 : 32 out-channels each
  const int wn    = wv & 1;              // 0..1 : 64 spatial each
  const int row16 = lane & 15;
  const int kb    = (lane < 16) ? 0 : 1; // K-half base in 8-f16 units

  // weight async-load thread mapping: 2 x 16B chunks per thread per tile
  //   chunk = j*256 + t; o = chunk>>2; c8 = chunk&3
  const int wo0 = t >> 2,           wc0 = t & 3;
  const int wo1 = (256 + t) >> 2,   wc1 = (256 + t) & 3;
  const unsigned wlds_base[2] = { (unsigned)(uintptr_t)&lds_wt[0][0],
                                  (unsigned)(uintptr_t)&lds_wt[1][0] };

  v8f acc[2][4];
  {
    const v8f z = {0.f, 0.f, 0.f, 0.f, 0.f, 0.f, 0.f, 0.f};
    #pragma unroll
    for (int i = 0; i < 2; ++i)
      #pragma unroll
      for (int j = 0; j < 4; ++j) acc[i][j] = z;
  }

  // ---- pipeline prologue: chunk 0 ----
  {
    const int kk = 0, cc = 0;
    async_load_b128_to_lds(wlds_base[0] + (unsigned)(wo0 * 5 + wc0) * 16,
                           wt + ((size_t)(kk * 128 + wo0) * 128) + cc * 32 + wc0 * 8);
    async_load_b128_to_lds(wlds_base[0] + (unsigned)(wo1 * 5 + wc1) * 16,
                           wt + ((size_t)(kk * 128 + wo1) * 128) + cc * 32 + wc1 * 8);
  }
  float aregs[16];
  load_act_chunk<SECOND>(0, n0, h0, w0, cgrp, in_f32, in_f16, aregs);

  for (int i = 0; i < NCHUNK; ++i) {
    const int cur = i & 1;
    // commit staged activation chunk i into LDS: 32B contiguous -> 2 x ds_store_b128
    {
      PackH pk;
      #pragma unroll
      for (int j = 0; j < 16; ++j) pk.h[j] = (_Float16)aregs[j];
      f32x4* dst = (f32x4*)&lds_act[cur][pos * LDSPITCH + cgrp * 16];
      dst[0] = pk.q[0];
      dst[1] = pk.q[1];
    }
    wait_asynccnt0();   // weight tile for chunk i has landed in lds_wt[cur]
    __syncthreads();

    // prefetch chunk i+1 while WMMAs below execute
    if (i + 1 < NCHUNK) {
      const int kk = (i + 1) >> 2, cc = (i + 1) & 3;
      const unsigned wb = wlds_base[cur ^ 1];
      async_load_b128_to_lds(wb + (unsigned)(wo0 * 5 + wc0) * 16,
                             wt + ((size_t)(kk * 128 + wo0) * 128) + cc * 32 + wc0 * 8);
      async_load_b128_to_lds(wb + (unsigned)(wo1 * 5 + wc1) * 16,
                             wt + ((size_t)(kk * 128 + wo1) * 128) + cc * 32 + wc1 * 8);
      load_act_chunk<SECOND>(i + 1, n0, h0, w0, cgrp, in_f32, in_f16, aregs);
    }

    const f32x4* wt4  = (const f32x4*)&lds_wt[cur][0];
    const f32x4* act4 = (const f32x4*)&lds_act[cur][0];
    FragH afr[2], bfr[4];
    #pragma unroll
    for (int fm = 0; fm < 2; ++fm) {
      const int r = wm * 32 + fm * 16 + row16;
      afr[fm].q[0] = wt4[r * 5 + kb];
      afr[fm].q[1] = wt4[r * 5 + kb + 2];
    }
    #pragma unroll
    for (int fn = 0; fn < 4; ++fn) {
      const int r = wn * 64 + fn * 16 + row16;
      bfr[fn].q[0] = act4[r * 5 + kb];
      bfr[fn].q[1] = act4[r * 5 + kb + 2];
    }
    #pragma unroll
    for (int fm = 0; fm < 2; ++fm)
      #pragma unroll
      for (int fn = 0; fn < 4; ++fn)
        acc[fm][fn] = __builtin_amdgcn_wmma_f32_16x16x32_f16(
            false, afr[fm].v, false, bfr[fn].v, (short)0, acc[fm][fn], false, false);
  }

  // epilogue: D layout -> lane holds N=spatial (row16), VGPR r holds M=o (+8 for hi half)
  const int ohi = (lane >= 16) ? 8 : 0;
  #pragma unroll
  for (int fm = 0; fm < 2; ++fm) {
    #pragma unroll
    for (int fn = 0; fn < 4; ++fn) {
      const int m = tile_base + wn * 64 + fn * 16 + row16;
      const int n = m / HWSZ;
      const int hw = m - n * HWSZ;
      #pragma unroll
      for (int r = 0; r < 8; ++r) {
        const int o = wm * 32 + fm * 16 + ohi + r;
        float v = acc[fm][fn][r];
        v = v * p_scale[o] + p_bias[o];                 // folded BN
        const float al = p_alpha[o];
        const float cl = fminf(fmaxf(v, 0.f), al);      // PACT clip
        const float q  = rintf(cl * p_qinv[o]) * p_qs[o];
        const size_t oidx = ((size_t)(n * CH + o)) * HWSZ + hw;
        if (!SECOND) {
          out_f16[oidx] = (_Float16)q;
        } else {
          const float s3 = p3_qs[o], i3 = p3_qinv[o];
          const float xv = xres[oidx];
          const float y  = rintf(q * i3) * s3 + rintf(xv * i3) * s3;
          const float c3 = fminf(fmaxf(y, 0.f), p3_alpha[o]);
          out_f32[oidx]  = rintf(c3 * i3) * s3;
        }
      }
    }
  }
}

extern "C" void kernel_launch(void* const* d_in, const int* in_sizes, int n_in,
                              void* d_out, int out_size, void* d_ws, size_t ws_size,
                              hipStream_t stream) {
  const float* x  = (const float*)d_in[0];
  const float* w1 = (const float*)d_in[1];
  const float* g1 = (const float*)d_in[2];
  const float* b1 = (const float*)d_in[3];
  const float* m1 = (const float*)d_in[4];
  const float* v1 = (const float*)d_in[5];
  const float* a1 = (const float*)d_in[6];
  const float* w2 = (const float*)d_in[7];
  const float* g2 = (const float*)d_in[8];
  const float* b2 = (const float*)d_in[9];
  const float* m2 = (const float*)d_in[10];
  const float* v2 = (const float*)d_in[11];
  const float* a2 = (const float*)d_in[12];
  const float* a3 = (const float*)d_in[13];

  const size_t WT = (size_t)KSZ * CH * CH;        // 147456 f16 per weight tensor
  _Float16* w1t = (_Float16*)d_ws;
  _Float16* w2t = w1t + WT;
  _Float16* mid = w2t + WT;                       // [N][C][H][W] f16, 25.7 MB

  wprep_kernel<<<dim3(256), dim3(128), 0, stream>>>(w1, w2, w1t, w2t);

  conv_kernel<false><<<dim3(MTOT / 128), dim3(256), 0, stream>>>(
      x, nullptr, w1t, g1, b1, m1, v1, a1, nullptr, nullptr, mid, nullptr);

  conv_kernel<true><<<dim3(MTOT / 128), dim3(256), 0, stream>>>(
      nullptr, mid, w2t, g2, b2, m2, v2, a2, a3, x, nullptr, (float*)d_out);
}
